// NetVariableClasses_57337813401735
// MI455X (gfx1250) — matrix operand order repaired and checked by reading, not compiled, not run on hardware
//
#include <hip/hip_runtime.h>
#include <stdint.h>

typedef __attribute__((ext_vector_type(16))) __bf16 v16bf;
typedef __attribute__((ext_vector_type(8)))  float  v8f;

#define N_NODES 50000
#define N_EDGES 250000
#define HIDC 128

__device__ __forceinline__ unsigned short f2bf(float f) {
  unsigned u = __float_as_uint(f);
  return (unsigned short)((u + 0x7FFFu + ((u >> 16) & 1u)) >> 16);
}

// ---------------------------------------------------------------------------
// Edge MLP hidden layer: per-channel batch stats of t = ea @ mw1 + mb1
// ---------------------------------------------------------------------------
__global__ void edge_stats_kernel(const float* __restrict__ ea,
                                  const float* __restrict__ mw1,
                                  const float* __restrict__ mb1,
                                  float* __restrict__ stats, int E) {
  __shared__ float sea[64 * 16];
  const int c = threadIdx.x;
  float wcol[16];
#pragma unroll
  for (int k = 0; k < 16; ++k) wcol[k] = mw1[k * HIDC + c];
  const float bias = mb1[c];
  float s = 0.f, s2 = 0.f;
  for (int e0 = blockIdx.x * 64; e0 < E; e0 += gridDim.x * 64) {
    const int cnt = min(64, E - e0);
    __syncthreads();
    for (int idx = threadIdx.x; idx < cnt * 16; idx += 128)
      sea[idx] = ea[(size_t)e0 * 16 + idx];
    __syncthreads();
    for (int e = 0; e < cnt; ++e) {
      float t = bias;
#pragma unroll
      for (int k = 0; k < 16; ++k) t += sea[e * 16 + k] * wcol[k];
      s += t; s2 += t * t;
    }
  }
  atomicAdd(&stats[c], s);
  atomicAdd(&stats[HIDC + c], s2);
}

__global__ void edge_bn_finalize_kernel(const float* __restrict__ stats,
                                        const float* __restrict__ g,
                                        const float* __restrict__ b,
                                        float* __restrict__ sc,
                                        float* __restrict__ sh, float invE) {
  const int c = threadIdx.x;
  const float m = stats[c] * invE;
  const float v = stats[HIDC + c] * invE - m * m;
  const float s = g[c] * rsqrtf(v + 1e-5f);
  sc[c] = s;
  sh[c] = b[c] - m * s;
}

// ---------------------------------------------------------------------------
// Recompute t, apply BN + ReLU, emit h in bf16 [E,128] (coalesced via LDS).
// ---------------------------------------------------------------------------
__global__ void edge_hidden_kernel(const float* __restrict__ ea,
                                   const float* __restrict__ mw1,
                                   const float* __restrict__ mb1,
                                   const float* __restrict__ sc,
                                   const float* __restrict__ sh,
                                   unsigned short* __restrict__ hout, int E) {
  __shared__ float sea[64 * 16];
  __shared__ unsigned short sout[64 * HIDC];
  const int c = threadIdx.x;
  float wcol[16];
#pragma unroll
  for (int k = 0; k < 16; ++k) wcol[k] = mw1[k * HIDC + c];
  const float bias = mb1[c];
  const float scale = sc[c], shift = sh[c];
  for (int e0 = blockIdx.x * 64; e0 < E; e0 += gridDim.x * 64) {
    const int cnt = min(64, E - e0);
    __syncthreads();
    for (int idx = threadIdx.x; idx < cnt * 16; idx += 128)
      sea[idx] = ea[(size_t)e0 * 16 + idx];
    __syncthreads();
    for (int e = 0; e < cnt; ++e) {
      float t = bias;
#pragma unroll
      for (int k = 0; k < 16; ++k) t += sea[e * 16 + k] * wcol[k];
      sout[e * HIDC + c] = f2bf(fmaxf(t * scale + shift, 0.f));
    }
    __syncthreads();
    const uint32_t* so = (const uint32_t*)sout;
    uint32_t* go = (uint32_t*)hout;
    for (int idx = threadIdx.x; idx < cnt * 64; idx += 128)
      go[(size_t)e0 * 64 + idx] = so[idx];
  }
}

// mw2 [128,P] fp32 -> mw2t [P,128] bf16 (A matrix for WMMA)
__global__ void transpose_w2_kernel(const float* __restrict__ mw2,
                                    unsigned short* __restrict__ w2t, int P) {
  const int idx = blockIdx.x * 256 + threadIdx.x;
  if (idx >= P * HIDC) return;
  const int p = idx >> 7, k = idx & 127;
  w2t[idx] = f2bf(mw2[(size_t)k * P + p]);
}

// ---------------------------------------------------------------------------
// Fused NNConv edge kernel (32-edge tiles, double-buffered async h staging):
//   Wt[p,e] = relu( sum_k mw2t[p,k]*h[e,k] + mb2[p] )   (bf16 WMMA, f32 acc)
//   msg[e,o] = sum_i x[src[e],i] * W[e, i*out_c+o]
//   atomicAdd(aggr[dst[e], o], msg)
// Grid: (edge-tile stride, P-slices of 256). 256 threads = 8 waves.
// ---------------------------------------------------------------------------
__global__ __launch_bounds__(256) void nnconv_edge_kernel(
    const unsigned short* __restrict__ hbuf,  // [E,128] bf16 bits
    const unsigned short* __restrict__ w2t,   // [P,128] bf16 bits
    const float* __restrict__ mb2,            // [P]
    const float* __restrict__ xnode,          // [N,in_c]
    const long long* __restrict__ ei,         // [2,E]
    float* __restrict__ aggr,                 // [N,out_c]
    int E, int icl, int in_c, int out_c, int P) {
  __shared__ unsigned short smw[256 * 136];    // mw2t slice, row pad 8 bf16
  __shared__ unsigned short shh[2][32 * 136];  // double-buffered h tiles
  __shared__ float          swv[32 * 264];     // relu'd weights (f32)
  __shared__ float          sxs[32 * 32];      // gathered x[src]
  __shared__ float          smb2[256];         // bias slice
  __shared__ int ssrc[32];
  __shared__ int sdst[32];

  const int tid = threadIdx.x;
  const int pstart = blockIdx.y * 256;
  const int pcnt = min(256, P - pstart);
  const int ntiles = pcnt >> 4;                // P is a multiple of 16
  const int i0 = pstart / out_c;
  const int icnt = pcnt / out_c;
  const int ntTotal = (E + 31) >> 5;

  // Load weight + bias slice into LDS once per block
  {
    const uint32_t* src = (const uint32_t*)w2t + (size_t)pstart * 64;
    uint32_t* dst = (uint32_t*)smw;
    for (int idx = tid; idx < pcnt * 64; idx += 256) {
      const int p = idx >> 6, k2 = idx & 63;
      dst[p * 68 + k2] = src[(size_t)p * 64 + k2];
    }
    for (int idx = tid; idx < pcnt; idx += 256) smb2[idx] = mb2[pstart + idx];
  }

  const int wave = tid >> 5;
  const int lane = tid & 31;
  const int l16 = lane & 15;
  const int lhalf = lane >> 4;

  // Async prefetch of one 32-edge h tile (32 rows x 256B) into buffer b.
  auto prefetch = [&](int t, int b) {
    const int eb = t << 5;
    for (int c = tid; c < 512; c += 256) {
      const int row = c >> 4, col = c & 15;
      const int erow = min(eb + row, E - 1);
      const char* g = (const char*)hbuf + (size_t)erow * 256 + (size_t)col * 16;
      const uint32_t l =
          (uint32_t)(uintptr_t)(&shh[b][(size_t)row * 136 + (size_t)col * 8]);
      asm volatile("global_load_async_to_lds_b128 %0, %1, off"
                   :: "v"(l), "v"(g) : "memory");
    }
  };

  int tile = blockIdx.x;
  int buf = 0;
  if (tile < ntTotal) prefetch(tile, 0);

  for (; tile < ntTotal; tile += gridDim.x, buf ^= 1) {
    const int e0 = tile << 5;
    const int cnt = min(32, E - e0);
    asm volatile("s_wait_asynccnt 0x0" ::: "memory");  // buf[cur] resident
    if (tid < cnt) ssrc[tid] = (int)ei[e0 + tid];
    if (tid >= 32 && tid - 32 < cnt)
      sdst[tid - 32] = (int)ei[(size_t)E + e0 + (tid - 32)];
    __syncthreads();  // ssrc visible; prev-iter readers of buf^1 done

    const int ntile = tile + gridDim.x;
    if (ntile < ntTotal) prefetch(ntile, buf ^ 1);  // overlap DMA w/ compute

    {  // gather x[src]
      const int total = cnt << icl;
      for (int idx = tid; idx < total; idx += 256) {
        const int e = idx >> icl, i = idx & (in_c - 1);
        sxs[e * 32 + i] = xnode[(size_t)ssrc[e] * in_c + i];
      }
    }
    __syncthreads();

    // WMMA: D = A(mw2t 16x32) x B(h^T 32x16) over K=128; 2 edge halves
    for (int t = wave; t < ntiles; t += 8) {
      const int p0l = t * 16;
      v8f acc0 = {0.f, 0.f, 0.f, 0.f, 0.f, 0.f, 0.f, 0.f};
      v8f acc1 = {0.f, 0.f, 0.f, 0.f, 0.f, 0.f, 0.f, 0.f};
      const unsigned short* arow = smw + (size_t)(p0l + l16) * 136;
      const unsigned short* b0r = &shh[buf][(size_t)l16 * 136];
      const unsigned short* b1r = &shh[buf][(size_t)(l16 + 16) * 136];
#pragma unroll
      for (int ks = 0; ks < 4; ++ks) {
        const int kb = ks * 32;
        union { v16bf v; uint4 q[2]; } A, B0, B1;
        // A lane layout: K = lhalf*16 .. +15 contiguous
        A.q[0] = *(const uint4*)(arow + kb + lhalf * 16);
        A.q[1] = *(const uint4*)(arow + kb + lhalf * 16 + 8);
        // B lane layout: K = lhalf*16 .. +15 contiguous
        const unsigned short* bp0 = b0r + kb + lhalf * 16;
        const unsigned short* bp1 = b1r + kb + lhalf * 16;
        B0.q[0] = *(const uint4*)bp0;
        B0.q[1] = *(const uint4*)(bp0 + 8);
        B1.q[0] = *(const uint4*)bp1;
        B1.q[1] = *(const uint4*)(bp1 + 8);
        acc0 = __builtin_amdgcn_wmma_f32_16x16x32_bf16(
            false, A.v, false, B0.v, (short)0, acc0, false, false);
        acc1 = __builtin_amdgcn_wmma_f32_16x16x32_bf16(
            false, A.v, false, B1.v, (short)0, acc1, false, false);
      }
      // D: lane holds e = l16 (+16), p = p0l + lhalf*8 + r
      const int pl = p0l + lhalf * 8;
      float* w0 = swv + (size_t)l16 * 264 + pl;
      float* w1 = swv + (size_t)(l16 + 16) * 264 + pl;
#pragma unroll
      for (int r = 0; r < 8; ++r) {
        const float bv = smb2[pl + r];
        w0[r] = fmaxf(acc0[r] + bv, 0.f);
        w1[r] = fmaxf(acc1[r] + bv, 0.f);
      }
    }
    __syncthreads();

    // per-edge matvec over this block's i-range + scatter-add
    for (int idx = tid; idx < 32 * out_c; idx += 256) {
      const int e = idx / out_c;
      if (e >= cnt) continue;
      const int o = idx - e * out_c;
      const float* we = swv + (size_t)e * 264;
      const float* xe = sxs + (size_t)e * 32 + i0;
      float s = 0.f;
      for (int i = 0; i < icnt; ++i) s += xe[i] * we[i * out_c + o];
      atomicAdd(&aggr[(size_t)sdst[e] * out_c + o], s);
    }
    __syncthreads();  // swv/sxs/ssrc reusable next iteration
  }
}

// ---------------------------------------------------------------------------
// Node side: y = aggr + x @ root + bias  (+ BN batch stats)
// ---------------------------------------------------------------------------
template <int IC>
__global__ void node_combine_kernel(const float* __restrict__ aggr,
                                    const float* __restrict__ xn,
                                    const float* __restrict__ root,
                                    const float* __restrict__ bias,
                                    float* __restrict__ y,
                                    float* __restrict__ nstats,
                                    int N, int out_c, int doStats) {
  __shared__ float sroot[IC * 32];
  __shared__ float sbias[32];
  __shared__ float ssum[32], ssq[32];
  for (int i = threadIdx.x; i < IC * out_c; i += blockDim.x) sroot[i] = root[i];
  if (threadIdx.x < 32) { ssum[threadIdx.x] = 0.f; ssq[threadIdx.x] = 0.f; }
  if (threadIdx.x < out_c) sbias[threadIdx.x] = bias[threadIdx.x];
  __syncthreads();
  for (int n = blockIdx.x * blockDim.x + threadIdx.x; n < N;
       n += gridDim.x * blockDim.x) {
    float xv[IC];
#pragma unroll
    for (int i = 0; i < IC; ++i) xv[i] = xn[(size_t)n * IC + i];
    for (int o = 0; o < out_c; ++o) {
      float v = sbias[o] + aggr[(size_t)n * out_c + o];
#pragma unroll
      for (int i = 0; i < IC; ++i) v += xv[i] * sroot[i * out_c + o];
      y[(size_t)n * out_c + o] = v;
      if (doStats) { atomicAdd(&ssum[o], v); atomicAdd(&ssq[o], v * v); }
    }
  }
  __syncthreads();
  if (doStats && threadIdx.x < out_c) {
    atomicAdd(&nstats[threadIdx.x], ssum[threadIdx.x]);
    atomicAdd(&nstats[32 + threadIdx.x], ssq[threadIdx.x]);
  }
}

__global__ void node_bn_finalize_kernel(const float* __restrict__ stats,
                                        const float* __restrict__ g,
                                        const float* __restrict__ b,
                                        float* __restrict__ sc,
                                        float* __restrict__ sh,
                                        int C, float invN) {
  const int c = threadIdx.x;
  if (c >= C) return;
  const float m = stats[c] * invN;
  const float v = stats[32 + c] * invN - m * m;
  const float s = g[c] * rsqrtf(v + 1e-5f);
  sc[c] = s;
  sh[c] = b[c] - m * s;
}

// h = relu(concat([bn(y), x]))
__global__ void apply_conv1_kernel(const float* __restrict__ y,
                                   const float* __restrict__ x,
                                   const float* __restrict__ sc,
                                   const float* __restrict__ sh,
                                   float* __restrict__ hout, int N) {
  const int n = blockIdx.x * 256 + threadIdx.x;
  if (n >= N) return;
#pragma unroll
  for (int o = 0; o < 28; ++o)
    hout[(size_t)n * 32 + o] = fmaxf(y[(size_t)n * 28 + o] * sc[o] + sh[o], 0.f);
#pragma unroll
  for (int j = 0; j < 4; ++j)
    hout[(size_t)n * 32 + 28 + j] = fmaxf(x[(size_t)n * 4 + j], 0.f);
}

// h = relu(bn(y) + skip)
__global__ void apply_res_kernel(const float* __restrict__ y,
                                 const float* __restrict__ hin,
                                 const float* __restrict__ sc,
                                 const float* __restrict__ sh,
                                 float* __restrict__ hout, int total) {
  const int idx = blockIdx.x * 256 + threadIdx.x;
  if (idx >= total) return;
  const int o = idx & 31;
  hout[idx] = fmaxf(y[idx] * sc[o] + sh[o] + hin[idx], 0.f);
}

// out[0:N] = clip(sigmoid(logit)); out[N:2N] = logit
__global__ void apply_final_kernel(const float* __restrict__ y,
                                   float* __restrict__ out, int N) {
  const int n = blockIdx.x * 256 + threadIdx.x;
  if (n >= N) return;
  const float l = y[n];
  float s = 1.f / (1.f + expf(-l));
  s = fminf(fmaxf(s, 1e-6f), 0.999999f);
  out[n] = s;
  out[N + n] = l;
}

// ---------------------------------------------------------------------------
extern "C" void kernel_launch(void* const* d_in, const int* in_sizes, int n_in,
                              void* d_out, int out_size, void* d_ws,
                              size_t ws_size, hipStream_t stream) {
  (void)in_sizes; (void)n_in; (void)out_size; (void)ws_size;
  const int N = N_NODES, E = N_EDGES;
  const float* x = (const float*)d_in[0];
  const long long* ei = (const long long*)d_in[1];
  const float* ea = (const float*)d_in[2];
  auto cp = [&](int ci, int j) { return (const float*)d_in[3 + ci * 8 + j]; };

  char* wp = (char*)d_ws;
  auto carve = [&](size_t bytes) {
    void* p = (void*)wp;
    wp += (bytes + 255) & ~(size_t)255;
    return p;
  };
  unsigned short* hbuf = (unsigned short*)carve((size_t)E * 128 * 2);
  unsigned short* w2t  = (unsigned short*)carve((size_t)1024 * 128 * 2);
  float* aggr   = (float*)carve((size_t)N * 32 * 4);
  float* ybuf   = (float*)carve((size_t)N * 32 * 4);
  float* hA     = (float*)carve((size_t)N * 32 * 4);
  float* hB     = (float*)carve((size_t)N * 32 * 4);
  float* estats = (float*)carve(256 * 4);
  float* esc    = (float*)carve(128 * 4);
  float* esh    = (float*)carve(128 * 4);
  float* nstats = (float*)carve(64 * 4);
  float* nsc    = (float*)carve(32 * 4);
  float* nsh    = (float*)carve(32 * 4);

  auto run_conv = [&](int ci, const float* xnode, int in_c, int out_c,
                      const float* bg, const float* bb, int doBN) {
    const float* mw1 = cp(ci, 0); const float* mb1 = cp(ci, 1);
    const float* mg  = cp(ci, 2); const float* mbt = cp(ci, 3);
    const float* mw2 = cp(ci, 4); const float* mb2 = cp(ci, 5);
    const float* root = cp(ci, 6); const float* bias = cp(ci, 7);
    const int P = in_c * out_c;
    (void)hipMemsetAsync(estats, 0, 256 * 4, stream);
    (void)hipMemsetAsync(aggr, 0, (size_t)N * out_c * 4, stream);
    if (doBN) (void)hipMemsetAsync(nstats, 0, 64 * 4, stream);
    edge_stats_kernel<<<512, 128, 0, stream>>>(ea, mw1, mb1, estats, E);
    edge_bn_finalize_kernel<<<1, 128, 0, stream>>>(estats, mg, mbt, esc, esh,
                                                   1.f / (float)E);
    edge_hidden_kernel<<<512, 128, 0, stream>>>(ea, mw1, mb1, esc, esh, hbuf, E);
    transpose_w2_kernel<<<(P * 128 + 255) / 256, 256, 0, stream>>>(mw2, w2t, P);
    const int icl = (in_c == 4) ? 2 : 5;
    dim3 grid(256, (unsigned)((P + 255) / 256));
    nnconv_edge_kernel<<<grid, 256, 0, stream>>>(hbuf, w2t, mb2, xnode, ei,
                                                 aggr, E, icl, in_c, out_c, P);
    if (in_c == 4)
      node_combine_kernel<4><<<256, 256, 0, stream>>>(aggr, xnode, root, bias,
                                                      ybuf, nstats, N, out_c,
                                                      doBN);
    else
      node_combine_kernel<32><<<256, 256, 0, stream>>>(aggr, xnode, root, bias,
                                                       ybuf, nstats, N, out_c,
                                                       doBN);
    if (doBN)
      node_bn_finalize_kernel<<<1, 32, 0, stream>>>(nstats, bg, bb, nsc, nsh,
                                                    out_c, 1.f / (float)N);
  };

  // conv1: 4 -> 28, bn1, relu(concat)
  run_conv(0, x, 4, 28, (const float*)d_in[43], (const float*)d_in[44], 1);
  apply_conv1_kernel<<<(N + 255) / 256, 256, 0, stream>>>(ybuf, x, nsc, nsh,
                                                          hA, N);
  // conv7/8/9: 32 -> 32, bn + residual relu
  run_conv(1, hA, 32, 32, (const float*)d_in[45], (const float*)d_in[46], 1);
  apply_res_kernel<<<(N * 32 + 255) / 256, 256, 0, stream>>>(ybuf, hA, nsc, nsh,
                                                             hB, N * 32);
  run_conv(2, hB, 32, 32, (const float*)d_in[47], (const float*)d_in[48], 1);
  apply_res_kernel<<<(N * 32 + 255) / 256, 256, 0, stream>>>(ybuf, hB, nsc, nsh,
                                                             hA, N * 32);
  run_conv(3, hA, 32, 32, (const float*)d_in[49], (const float*)d_in[50], 1);
  apply_res_kernel<<<(N * 32 + 255) / 256, 256, 0, stream>>>(ybuf, hA, nsc, nsh,
                                                             hB, N * 32);
  // conv10: 32 -> 1, no BN
  run_conv(4, hB, 32, 1, nullptr, nullptr, 0);
  apply_final_kernel<<<(N + 255) / 256, 256, 0, stream>>>(ybuf, (float*)d_out,
                                                          N);
}
